// Attention_5849745457264
// MI455X (gfx1250) — matrix operand order repaired and checked by reading, not compiled, not run on hardware
//
#include <hip/hip_runtime.h>
#include <hip/hip_bf16.h>

// ---------------------------------------------------------------------------
// Fused multi-head attention for MI455X (gfx1250, wave32, WMMA).
//   1) cvt:   fp32 -> f16 staging of x, w_qkv, w_o
//   2) qkv:   GEMM M=8192 N=1536 K=512 -> Q (pre-scaled 0.125), K, V^T (f16)
//             64x64 block per wave: B-fragment reused across 4 A-fragments.
//   3) attn:  flash attention; 4-wave WG, 128 queries/WG; K/V tiles staged
//             to LDS with double-buffered async copies (ASYNCcnt path).
//   4) oproj: GEMM M=8192 N=512 K=512 + bias -> fp32 (B,N,D)
// All matrix math: v_wmma_f32_16x16x32_f16 (f32 accumulation).
// ---------------------------------------------------------------------------

typedef __attribute__((ext_vector_type(16))) _Float16 v16h;
typedef __attribute__((ext_vector_type(8)))  _Float16 v8h;
typedef __attribute__((ext_vector_type(8)))  float    v8f;
typedef __attribute__((ext_vector_type(4)))  int      v4i;

#define NSEQ   4096
#define DMODEL 512
#define HEADS  8
#define HD     64
#define NTOK   8192   // N*B

// LDS strides (f16 elements): 16B-aligned chunk starts, bank-spread rows.
#define KLD 72   // 32 keys x 64 hd tile
#define VLD 40   // 64 hd  x 32 key tile
#define PLD 40   // 32 q   x 32 key P tile (per wave)

// --- gfx1250 async global->LDS copy (ASYNCcnt), with safe fallback ---------
#if defined(__has_builtin)
#if __has_builtin(__builtin_amdgcn_global_load_async_to_lds_b128)
#define ASYNC_COPY 1
#endif
#endif
#ifndef ASYNC_COPY
#define ASYNC_COPY 0
#endif

#define AS3 __attribute__((address_space(3)))

#if ASYNC_COPY
#if __has_builtin(__builtin_amdgcn_s_wait_asynccnt)
#define WAIT_ASYNC() __builtin_amdgcn_s_wait_asynccnt(0)
#else
#define WAIT_ASYNC() asm volatile("s_wait_asynccnt 0x0" ::: "memory")
#endif
// param0: int4* (per clang-22 diagnostic), param1: LDS pointer (as3)
#define COPY16B(gsrc, ldst)                                                  \
  __builtin_amdgcn_global_load_async_to_lds_b128(                            \
      (v4i*)(gsrc), (AS3 v4i*)(ldst), 0, 0)
#else
#define WAIT_ASYNC()
#define COPY16B(gsrc, ldst) *(v8h*)(ldst) = *(const v8h*)(gsrc)
#endif

static __device__ __forceinline__ v8f vzero8() {
  v8f v;
#pragma unroll
  for (int i = 0; i < 8; ++i) v[i] = 0.0f;
  return v;
}

static __device__ __forceinline__ v16h cat8(v8h lo, v8h hi) {
  return __builtin_shufflevector(lo, hi, 0, 1, 2, 3, 4, 5, 6, 7,
                                         8, 9, 10, 11, 12, 13, 14, 15);
}

// A fragment (16x32 f16, MxK) from row-major [row][k], leading dim ld.
// Lanes 0-15: row M=lane, K-chunks {0..7,16..23}; lanes 16-31: {8..15,24..31}.
static __device__ __forceinline__ v16h load_a(const _Float16* base, int ld,
                                              int row0, int k0, int lane) {
  int r = lane & 15, hf = lane >> 4;
  const _Float16* p = base + (size_t)(row0 + r) * ld + (k0 + hf * 8);
  v8h lo = *(const v8h*)p;
  v8h hi = *(const v8h*)(p + 16);
  return cat8(lo, hi);
}

// B fragment (32x16 f16, KxN): column n of B is row (col0+n) of `base`,
// lanes 0-15: K-chunks {0..7,16..23}; lanes 16-31: {8..15,24..31}.
static __device__ __forceinline__ v16h load_b(const _Float16* base, int ld,
                                              int col0, int k0, int lane) {
  int c = lane & 15, hf = lane >> 4;
  const _Float16* p = base + (size_t)(col0 + c) * ld + (k0 + hf * 8);
  v8h lo = *(const v8h*)p;
  v8h hi = *(const v8h*)(p + 16);
  return cat8(lo, hi);
}

static __device__ __forceinline__ v8f wmma16(v16h a, v16h b, v8f c) {
  return __builtin_amdgcn_wmma_f32_16x16x32_f16(false, a, false, b,
                                                (short)0, c, false, false);
}

// ---------------------------------------------------------------------------
__global__ void cvt_f16_kernel(const float* __restrict__ src,
                               _Float16* __restrict__ dst, int n) {
  int i = blockIdx.x * blockDim.x + threadIdx.x;
  int stride = gridDim.x * blockDim.x;
  for (; i < n; i += stride) dst[i] = (_Float16)src[i];
}

// ---------------------------------------------------------------------------
// QKV projection: qkv[t][e] = sum_d x[t][d] * w_qkv[e][d].
// One wave computes a 64x64 block (16 accumulator tiles).
__global__ __launch_bounds__(32) void qkv_kernel(
    const _Float16* __restrict__ X,    // [8192][512]
    const _Float16* __restrict__ W,    // [1536][512]
    _Float16* __restrict__ Qb,         // [16][4096][64]
    _Float16* __restrict__ Kb,         // [16][4096][64]
    _Float16* __restrict__ VT) {       // [16][64][4096]
  int lane = threadIdx.x;
  int m0 = blockIdx.x * 64;
  int e0 = blockIdx.y * 64;
  v8f c[4][4];
#pragma unroll
  for (int mt = 0; mt < 4; ++mt)
#pragma unroll
    for (int nt = 0; nt < 4; ++nt) c[mt][nt] = vzero8();

  for (int k = 0; k < DMODEL; k += 32) {
    v16h a0 = load_a(X, DMODEL, m0 +  0, k, lane);
    v16h a1 = load_a(X, DMODEL, m0 + 16, k, lane);
    v16h a2 = load_a(X, DMODEL, m0 + 32, k, lane);
    v16h a3 = load_a(X, DMODEL, m0 + 48, k, lane);
#pragma unroll
    for (int nt = 0; nt < 4; ++nt) {
      v16h bf = load_b(W, DMODEL, e0 + nt * 16, k, lane);
      c[0][nt] = wmma16(a0, bf, c[0][nt]);
      c[1][nt] = wmma16(a1, bf, c[1][nt]);
      c[2][nt] = wmma16(a2, bf, c[2][nt]);
      c[3][nt] = wmma16(a3, bf, c[3][nt]);
    }
  }
  int sel = e0 >> 9;           // 0:q 1:k 2:v
  int h   = (e0 >> 6) & 7;     // head
  int rr = lane & 15, hh = lane >> 4;
#pragma unroll
  for (int mt = 0; mt < 4; ++mt) {
#pragma unroll
    for (int r = 0; r < 8; ++r) {
      int tok = m0 + mt * 16 + r + 8 * hh;
      int n = tok >> 1, b = tok & 1;       // x layout is (N, B=2, D)
      int bh = b * HEADS + h;
      if (sel == 0) {
        size_t base = ((size_t)bh * NSEQ + n) * HD;
#pragma unroll
        for (int nt = 0; nt < 4; ++nt)
          Qb[base + nt * 16 + rr] = (_Float16)(c[mt][nt][r] * 0.125f);
      } else if (sel == 1) {
        size_t base = ((size_t)bh * NSEQ + n) * HD;
#pragma unroll
        for (int nt = 0; nt < 4; ++nt)
          Kb[base + nt * 16 + rr] = (_Float16)c[mt][nt][r];
      } else {
        size_t base = (size_t)bh * HD * NSEQ + n;
#pragma unroll
        for (int nt = 0; nt < 4; ++nt)
          VT[base + (size_t)(nt * 16 + rr) * NSEQ] = (_Float16)c[mt][nt][r];
      }
    }
  }
}

// ---------------------------------------------------------------------------
// Flash attention: 4 waves/WG, 128 queries/WG (32 per wave).  K/V tiles of
// 32 keys staged into LDS with double-buffered async copies; P transposed
// C->A layout through per-wave LDS scratch.
__global__ __launch_bounds__(128) void attn_kernel(
    const _Float16* __restrict__ Q,   // [16][4096][64] (pre-scaled)
    const _Float16* __restrict__ K,   // [16][4096][64]
    const _Float16* __restrict__ VT,  // [16][64][4096]
    _Float16* __restrict__ O) {       // [2][4096][512]
  __shared__ __align__(16) _Float16 kbuf[2][32 * KLD];
  __shared__ __align__(16) _Float16 vbuf[2][64 * VLD];
  __shared__ __align__(16) _Float16 pbuf[4][32 * PLD];

  int tid = threadIdx.x;
  int lane = tid & 31, wv = tid >> 5;
  int bh = blockIdx.y;
  int b = bh >> 3, h = bh & 7;
  const _Float16* Qp = Q + (size_t)bh * NSEQ * HD;
  const _Float16* Kp = K + (size_t)bh * NSEQ * HD;
  const _Float16* Vp = VT + (size_t)bh * HD * NSEQ;
  int rr = lane & 15, hh = lane >> 4;
  int qw = blockIdx.x * 128 + wv * 32;   // this wave's 32 queries

  v16h qa[2][2];
#pragma unroll
  for (int qt = 0; qt < 2; ++qt) {
    qa[qt][0] = load_a(Qp, HD, qw + qt * 16, 0, lane);
    qa[qt][1] = load_a(Qp, HD, qw + qt * 16, 32, lane);
  }
  v8f acc[2][4];
  float mrun[2][8], lrun[2][8];
#pragma unroll
  for (int qt = 0; qt < 2; ++qt) {
#pragma unroll
    for (int t = 0; t < 4; ++t) acc[qt][t] = vzero8();
#pragma unroll
    for (int r = 0; r < 8; ++r) { mrun[qt][r] = -1e30f; lrun[qt][r] = 0.0f; }
  }

  // cooperative stage of one 32-key K tile + V^T tile (512 x 16B chunks)
  auto stage = [&](int buf, int j) {
#pragma unroll
    for (int i = 0; i < 2; ++i) {
      int ch = tid + i * 128;
      int kk = ch >> 3, kc = ch & 7;                      // K: 32r x 8 chunks
      COPY16B(Kp + (size_t)(j + kk) * HD + kc * 8, &kbuf[buf][kk * KLD + kc * 8]);
      int vr = ch >> 2, vc = ch & 3;                      // V: 64r x 4 chunks
      COPY16B(Vp + (size_t)vr * NSEQ + j + vc * 8, &vbuf[buf][vr * VLD + vc * 8]);
    }
  };

  stage(0, 0);
  int cur = 0;
  for (int j = 0; j < NSEQ; j += 32) {
    WAIT_ASYNC();
    __syncthreads();                       // tile `cur` visible to all waves
    if (j + 32 < NSEQ) stage(cur ^ 1, j + 32);
    const _Float16* kb = kbuf[cur];
    const _Float16* vb = vbuf[cur];
    _Float16* pb = pbuf[wv];

#pragma unroll
    for (int qt = 0; qt < 2; ++qt) {
      v8f s0 = vzero8(), s1 = vzero8();
      s0 = wmma16(qa[qt][0], load_b(kb, KLD,  0,  0, lane), s0);
      s0 = wmma16(qa[qt][1], load_b(kb, KLD,  0, 32, lane), s0);
      s1 = wmma16(qa[qt][0], load_b(kb, KLD, 16,  0, lane), s1);
      s1 = wmma16(qa[qt][1], load_b(kb, KLD, 16, 32, lane), s1);
      // online softmax per row (M = r + 8*hh, col = lane%16)
#pragma unroll
      for (int r = 0; r < 8; ++r) {
        float mx = fmaxf(s0[r], s1[r]);
        for (int m = 1; m < 16; m <<= 1) mx = fmaxf(mx, __shfl_xor(mx, m, 32));
        float mnew  = fmaxf(mrun[qt][r], mx);
        float alpha = __expf(mrun[qt][r] - mnew);
        float p0 = __expf(s0[r] - mnew);
        float p1 = __expf(s1[r] - mnew);
        float rs = p0 + p1;
        for (int m = 1; m < 16; m <<= 1) rs += __shfl_xor(rs, m, 32);
        lrun[qt][r] = lrun[qt][r] * alpha + rs;
        mrun[qt][r] = mnew;
        acc[qt][0][r] *= alpha; acc[qt][1][r] *= alpha;
        acc[qt][2][r] *= alpha; acc[qt][3][r] *= alpha;
        int row = qt * 16 + r + 8 * hh;
        pb[row * PLD + rr]      = (_Float16)p0;
        pb[row * PLD + 16 + rr] = (_Float16)p1;
      }
    }
    // P @ V^T   (same-wave LDS in-order: P writes above feed these reads)
#pragma unroll
    for (int qt = 0; qt < 2; ++qt) {
      v16h pa = load_a(pb, PLD, qt * 16, 0, lane);
      acc[qt][0] = wmma16(pa, load_b(vb, VLD,  0, 0, lane), acc[qt][0]);
      acc[qt][1] = wmma16(pa, load_b(vb, VLD, 16, 0, lane), acc[qt][1]);
      acc[qt][2] = wmma16(pa, load_b(vb, VLD, 32, 0, lane), acc[qt][2]);
      acc[qt][3] = wmma16(pa, load_b(vb, VLD, 48, 0, lane), acc[qt][3]);
    }
    __syncthreads();                       // all reads of `cur` done
    cur ^= 1;
  }

#pragma unroll
  for (int qt = 0; qt < 2; ++qt) {
#pragma unroll
    for (int r = 0; r < 8; ++r) {
      float inv = 1.0f / lrun[qt][r];
      int row = qw + qt * 16 + r + 8 * hh;
      size_t base = ((size_t)b * NSEQ + row) * DMODEL + h * HD;
      O[base +  0 + rr] = (_Float16)(acc[qt][0][r] * inv);
      O[base + 16 + rr] = (_Float16)(acc[qt][1][r] * inv);
      O[base + 32 + rr] = (_Float16)(acc[qt][2][r] * inv);
      O[base + 48 + rr] = (_Float16)(acc[qt][3][r] * inv);
    }
  }
}

// ---------------------------------------------------------------------------
// Output projection: out[u][e] = sum_d O[u][d]*w_o[e][d] + b_o[e], fp32 out.
// One wave computes a 64x64 block.
__global__ __launch_bounds__(32) void oproj_kernel(
    const _Float16* __restrict__ O,   // [8192][512]
    const _Float16* __restrict__ W,   // [512][512]
    const float* __restrict__ bias,   // [512]
    float* __restrict__ out) {        // [2][4096][512] fp32
  int lane = threadIdx.x;
  int m0 = blockIdx.x * 64;
  int e0 = blockIdx.y * 64;
  v8f c[4][4];
#pragma unroll
  for (int mt = 0; mt < 4; ++mt)
#pragma unroll
    for (int nt = 0; nt < 4; ++nt) c[mt][nt] = vzero8();

  for (int k = 0; k < DMODEL; k += 32) {
    v16h a0 = load_a(O, DMODEL, m0 +  0, k, lane);
    v16h a1 = load_a(O, DMODEL, m0 + 16, k, lane);
    v16h a2 = load_a(O, DMODEL, m0 + 32, k, lane);
    v16h a3 = load_a(O, DMODEL, m0 + 48, k, lane);
#pragma unroll
    for (int nt = 0; nt < 4; ++nt) {
      v16h bf = load_b(W, DMODEL, e0 + nt * 16, k, lane);
      c[0][nt] = wmma16(a0, bf, c[0][nt]);
      c[1][nt] = wmma16(a1, bf, c[1][nt]);
      c[2][nt] = wmma16(a2, bf, c[2][nt]);
      c[3][nt] = wmma16(a3, bf, c[3][nt]);
    }
  }
  int rr = lane & 15, hh = lane >> 4;
#pragma unroll
  for (int mt = 0; mt < 4; ++mt) {
#pragma unroll
    for (int r = 0; r < 8; ++r) {
      size_t base = (size_t)(m0 + mt * 16 + r + 8 * hh) * DMODEL + e0;
#pragma unroll
      for (int nt = 0; nt < 4; ++nt)
        out[base + nt * 16 + rr] = c[mt][nt][r] + bias[e0 + nt * 16 + rr];
    }
  }
}

// ---------------------------------------------------------------------------
extern "C" void kernel_launch(void* const* d_in, const int* in_sizes, int n_in,
                              void* d_out, int out_size, void* d_ws,
                              size_t ws_size, hipStream_t stream) {
  (void)in_sizes; (void)n_in; (void)out_size; (void)ws_size;
  const float* x    = (const float*)d_in[0];  // (4096, 2, 512)
  const float* wqkv = (const float*)d_in[1];  // (1536, 512)
  const float* wo   = (const float*)d_in[2];  // (512, 512)
  const float* bo   = (const float*)d_in[3];  // (512,)
  float* out = (float*)d_out;

  char* ws = (char*)d_ws;
  const size_t SZ_X  = (size_t)NTOK * DMODEL * 2;          // 8 MiB
  const size_t SZ_WQ = (size_t)3 * DMODEL * DMODEL * 2;    // 1.5 MiB
  const size_t SZ_WO = (size_t)DMODEL * DMODEL * 2;        // 0.5 MiB
  const size_t SZ_QKV = (size_t)16 * NSEQ * HD * 2;        // 8 MiB each
  _Float16* xb  = (_Float16*)(ws);
  _Float16* wqb = (_Float16*)(ws + SZ_X);
  _Float16* wob = (_Float16*)(ws + SZ_X + SZ_WQ);
  _Float16* Qb  = (_Float16*)(ws + SZ_X + SZ_WQ + SZ_WO);
  _Float16* Kb  = (_Float16*)(ws + SZ_X + SZ_WQ + SZ_WO + SZ_QKV);
  _Float16* VT  = (_Float16*)(ws + SZ_X + SZ_WQ + SZ_WO + 2 * SZ_QKV);
  _Float16* Ob  = (_Float16*)(ws + SZ_X + SZ_WQ + SZ_WO + 3 * SZ_QKV);

  cvt_f16_kernel<<<2048, 256, 0, stream>>>(x, xb, NTOK * DMODEL);
  cvt_f16_kernel<<<512, 256, 0, stream>>>(wqkv, wqb, 3 * DMODEL * DMODEL);
  cvt_f16_kernel<<<256, 256, 0, stream>>>(wo, wob, DMODEL * DMODEL);

  qkv_kernel<<<dim3(NTOK / 64, (3 * DMODEL) / 64), 32, 0, stream>>>(
      xb, wqb, Qb, Kb, VT);

  attn_kernel<<<dim3(NSEQ / 128, 16), 128, 0, stream>>>(Qb, Kb, VT, Ob);

  oproj_kernel<<<dim3(NTOK / 64, DMODEL / 64), 32, 0, stream>>>(
      Ob, wob, bo, out);
}
